// LinearTemporalSelfAttention_64046552318268
// MI455X (gfx1250) — matrix operand order repaired and checked by reading, not compiled, not run on hardware
//
#include <hip/hip_runtime.h>

typedef __bf16 bf16;
typedef __attribute__((ext_vector_type(16))) __bf16 v16bf;
typedef __attribute__((ext_vector_type(8)))  float  v8f;
typedef __attribute__((ext_vector_type(4)))  unsigned int u32x4;
typedef __attribute__((ext_vector_type(8)))  int i32x8;

// Padded LDS row strides (bf16 elems), 16B-aligned rows, conflict-free b128 frag loads
#define LSTR  40   // for 32-deep K tiles (80B rows)
#define LSTR2 72   // for 64-deep K tiles (144B rows)

#ifndef USE_TDM
#define USE_TDM 1  // tensor_load_to_lds for the att tile in ygemm
#endif

union F16x16 { v16bf v; bf16 s[16]; uint4 q[2]; };

// A-fragment (16x32 bf16, ISA 7.12.2): lane holds M=lane&15; two contiguous K-runs.
__device__ __forceinline__ void load_a_frag(F16x16& f, const bf16* rowBase, int kh) {
  const uint4* p = (const uint4*)rowBase;   // row-major [m][k]
  f.q[0] = p[kh];        // K = 8*kh + 0..7
  f.q[1] = p[2 + kh];    // K = 16 + 8*kh + 0..7
}
// B-fragment (32x16 bf16): lane holds N=lane&15; K = 8*kh + 0..7, 16 + 8*kh + 0..7
// when LDS tile is stored transposed [n][k].
__device__ __forceinline__ void load_b_frag(F16x16& f, const bf16* rowBase, int kh) {
  const uint4* p = (const uint4*)rowBase;
  f.q[0] = p[kh];
  f.q[1] = p[2 + kh];
}

// ---- block reductions (blockDim == 256) ----
__device__ __forceinline__ float blockReduceSum(float v) {
  __shared__ float sm[8];
  int lane = threadIdx.x & 31, w = threadIdx.x >> 5;
#pragma unroll
  for (int o = 16; o > 0; o >>= 1) v += __shfl_xor(v, o, 32);
  if (lane == 0) sm[w] = v;
  __syncthreads();
  float r = (lane < 8) ? sm[lane] : 0.0f;
#pragma unroll
  for (int o = 4; o > 0; o >>= 1) r += __shfl_xor(r, o, 32);
  r = __shfl(r, 0, 32);
  __syncthreads();
  return r;
}
__device__ __forceinline__ float blockReduceMax(float v) {
  __shared__ float sm[8];
  int lane = threadIdx.x & 31, w = threadIdx.x >> 5;
#pragma unroll
  for (int o = 16; o > 0; o >>= 1) v = fmaxf(v, __shfl_xor(v, o, 32));
  if (lane == 0) sm[w] = v;
  __syncthreads();
  float r = (lane < 8) ? sm[lane] : -3.4e38f;
#pragma unroll
  for (int o = 4; o > 0; o >>= 1) r = fmaxf(r, __shfl_xor(r, o, 32));
  r = __shfl(r, 0, 32);
  __syncthreads();
  return r;
}

// ---- fp32 -> bf16 transposed convert: dst[c*rows + r] = src[r*cols + c] ----
__global__ void f32_to_bf16_T_kernel(const float* __restrict__ src, bf16* __restrict__ dst,
                                     int rows, int cols) {
  int n = rows * cols;
  for (int i = blockIdx.x * blockDim.x + threadIdx.x; i < n; i += gridDim.x * blockDim.x) {
    int r = i / cols, c = i - r * cols;
    dst[(size_t)c * rows + r] = (bf16)src[i];
  }
}

// ---- LayerNorm (row = 1024 floats, 256 threads) -> bf16 ----
__global__ __launch_bounds__(256) void layernorm_bf16_kernel(
    const float* __restrict__ X, const float* __restrict__ g, const float* __restrict__ bta,
    bf16* __restrict__ Out, int D) {
  size_t row = blockIdx.x;
  const float* xr = X + row * D;
  float vals[4];
  float s = 0.f;
#pragma unroll
  for (int i = 0; i < 4; i++) { vals[i] = xr[threadIdx.x + i * 256]; s += vals[i]; }
  float mean = blockReduceSum(s) * (1.0f / D);
  float sq = 0.f;
#pragma unroll
  for (int i = 0; i < 4; i++) { float d = vals[i] - mean; sq += d * d; }
  float var = blockReduceSum(sq) * (1.0f / D);
  float rstd = rsqrtf(var + 1e-5f);
#pragma unroll
  for (int i = 0; i < 4; i++) {
    int c = threadIdx.x + i * 256;
    Out[row * D + c] = (bf16)((vals[i] - mean) * rstd * g[c] + bta[c]);
  }
}

// ---- Tiled bf16 WMMA GEMM: C(MxN) = A(MxK) * B^T-supplied (BwT is N x K row-major) ----
// block tile 128x256, 8 waves (2M x 4N), wave tile 64x64 = 4x4 wmma tiles, K stage 32,
// double-buffered LDS with register staging: one barrier per K-step.
// EPI 0: out f32[gm*N+gn] = acc + bias
// EPI 1: out bf16 transposed (b,d,t): [(bb*N+gn)*Tspan + tt] = (acc + bias)*rowscale[gm]
// EPI 2: out f32[gm*N+gn] = Xres + acc + bias
template <int EPI>
__global__ __launch_bounds__(256) void gemm_bf16_kernel(
    const bf16* __restrict__ A, const bf16* __restrict__ BwT,
    const float* __restrict__ bias, void* __restrict__ OutP,
    const float* __restrict__ rowscale, const float* __restrict__ Xres,
    int M, int N, int K, int Tspan) {
  __shared__ __align__(16) bf16 As[2][128 * LSTR];
  __shared__ __align__(16) bf16 Bs[2][256 * LSTR];
  const int tid = threadIdx.x;
  const int lane = tid & 31;
  const int wave = tid >> 5;
  const int kh = (lane >= 16) ? 1 : 0;
  const int wm = (wave >> 2) * 64;   // 2 waves over M
  const int wn = (wave & 3) * 64;    // 4 waves over N
  const int m0 = blockIdx.y * 128;
  const int n0 = blockIdx.x * 256;
  const int sr = tid >> 2;           // staging row base (0..63)
  const int sc = (tid & 3) * 8;      // staging col (bf16 elems)
  v8f acc[4][4] = {};
  // prologue: stage tile kk=0 in registers
  uint4 ra[2], rb[4];
#pragma unroll
  for (int it = 0; it < 2; ++it)
    ra[it] = *(const uint4*)(A + (size_t)(m0 + sr + it * 64) * K + sc);
#pragma unroll
  for (int it = 0; it < 4; ++it)
    rb[it] = *(const uint4*)(BwT + (size_t)(n0 + sr + it * 64) * K + sc);
  int buf = 0;
  for (int kk = 0; kk < K; kk += 32) {
#pragma unroll
    for (int it = 0; it < 2; ++it)
      ((uint4*)(As[buf] + (sr + it * 64) * LSTR))[tid & 3] = ra[it];
#pragma unroll
    for (int it = 0; it < 4; ++it)
      ((uint4*)(Bs[buf] + (sr + it * 64) * LSTR))[tid & 3] = rb[it];
    __syncthreads();
    int kn = kk + 32;
    if (kn < K) {   // issue next tile's global loads; they overlap the WMMA burst
#pragma unroll
      for (int it = 0; it < 2; ++it)
        ra[it] = *(const uint4*)(A + (size_t)(m0 + sr + it * 64) * K + kn + sc);
#pragma unroll
      for (int it = 0; it < 4; ++it)
        rb[it] = *(const uint4*)(BwT + (size_t)(n0 + sr + it * 64) * K + kn + sc);
    }
    if (kk + 64 < K) {  // L2 prefetch two tiles ahead (global_prefetch_b8)
      __builtin_prefetch(A + (size_t)(m0 + sr) * K + kk + 64, 0, 0);
      __builtin_prefetch(BwT + (size_t)(n0 + sr) * K + kk + 64, 0, 0);
    }
    F16x16 af[4], bfr[4];
#pragma unroll
    for (int mt = 0; mt < 4; ++mt)
      load_a_frag(af[mt], As[buf] + (wm + mt * 16 + (lane & 15)) * LSTR, kh);
#pragma unroll
    for (int nt = 0; nt < 4; ++nt)
      load_b_frag(bfr[nt], Bs[buf] + (wn + nt * 16 + (lane & 15)) * LSTR, kh);
#pragma unroll
    for (int mt = 0; mt < 4; ++mt)
#pragma unroll
      for (int nt = 0; nt < 4; ++nt)
        acc[mt][nt] = __builtin_amdgcn_wmma_f32_16x16x32_bf16(
            false, af[mt].v, false, bfr[nt].v, (short)0, acc[mt][nt], false, false);
    buf ^= 1;
  }
#pragma unroll
  for (int mt = 0; mt < 4; ++mt)
#pragma unroll
    for (int nt = 0; nt < 4; ++nt) {
      int gn = n0 + wn + nt * 16 + (lane & 15);
      float bi = bias[gn];
#pragma unroll
      for (int r = 0; r < 8; ++r) {
        int gm = m0 + wm + mt * 16 + r + ((lane >= 16) ? 8 : 0);
        float v = acc[mt][nt][r] + bi;
        if (EPI == 0) {
          ((float*)OutP)[(size_t)gm * N + gn] = v;
        } else if (EPI == 1) {
          int bb = gm / Tspan, tt = gm - bb * Tspan;
          ((bf16*)OutP)[((size_t)(bb * N + gn)) * Tspan + tt] = (bf16)(v * rowscale[gm]);
        } else {
          size_t idx = (size_t)gm * N + gn;
          ((float*)OutP)[idx] = Xres[idx] + v;
        }
      }
    }
}

// ---- q softmax over head dim 64 (one wave per (token,head) slice) -> bf16 ----
__global__ __launch_bounds__(256) void q_softmax_kernel(const float* __restrict__ Q,
                                                        bf16* __restrict__ Out) {
  int slice = blockIdx.x * 8 + (threadIdx.x >> 5);
  int lane = threadIdx.x & 31;
  size_t base = (size_t)slice * 64;
  float v0 = Q[base + lane], v1 = Q[base + 32 + lane];
  float m = fmaxf(v0, v1);
#pragma unroll
  for (int o = 16; o > 0; o >>= 1) m = fmaxf(m, __shfl_xor(m, o, 32));
  float e0 = __expf(v0 - m), e1 = __expf(v1 - m);
  float s = e0 + e1;
#pragma unroll
  for (int o = 16; o > 0; o >>= 1) s += __shfl_xor(s, o, 32);
  float inv = 1.0f / s;
  Out[base + lane] = (bf16)(e0 * inv);
  Out[base + 32 + lane] = (bf16)(e1 * inv);
}

// ---- k softmax over time with mask bias; writes TIME-MAJOR kT[b,d,t] (coalesced) ----
__global__ __launch_bounds__(256) void time_softmax_kernel(
    const float* __restrict__ Kraw, const float* __restrict__ mask,
    bf16* __restrict__ OutT, int T, int D) {
  int col = blockIdx.x;                 // b*D + d
  int b = col / D, d = col - b * D;
  float vv[8];
  float mx = -3.4e38f;
#pragma unroll
  for (int i = 0; i < 8; ++i) {
    int n = threadIdx.x + i * 256;
    float v = Kraw[((size_t)(b * T + n)) * D + d] + (1.0f - mask[b * T + n]) * (-1000000.0f);
    vv[i] = v; mx = fmaxf(mx, v);
  }
  mx = blockReduceMax(mx);
  float s = 0.f;
#pragma unroll
  for (int i = 0; i < 8; ++i) { vv[i] = __expf(vv[i] - mx); s += vv[i]; }
  s = blockReduceSum(s);
  float inv = 1.0f / s;
#pragma unroll
  for (int i = 0; i < 8; ++i) {
    int n = threadIdx.x + i * 256;
    OutT[(size_t)col * T + n] = (bf16)(vv[i] * inv);
  }
}

// ---- att^T[b,h][l][d] = (k^T v)^T over T; double-buffered, 1 barrier per step ----
__global__ __launch_bounds__(256) void attn_state_kernel(
    const bf16* __restrict__ kT, const bf16* __restrict__ vT,
    bf16* __restrict__ AttT, int T, int D, int H) {
  __shared__ __align__(16) bf16 Ks[2][64 * LSTR];   // [d][t] tile
  __shared__ __align__(16) bf16 Vs[2][64 * LSTR];   // [l][t] tile
  int bh = blockIdx.x;
  int b = bh / H, h = bh - b * H;
  int tid = threadIdx.x, lane = tid & 31, wave = tid >> 5;
  const int kh = (lane >= 16) ? 1 : 0;
  int mt = (wave & 3) * 16;
  int ntb = (wave >> 2) * 32;
  int r = tid >> 2, c = (tid & 3) * 8;
  size_t rowbase = ((size_t)(b * D + h * 64 + r)) * T;
  uint4 rk = *(const uint4*)(kT + rowbase + c);
  uint4 rv = *(const uint4*)(vT + rowbase + c);
  v8f acc[2] = {};
  int buf = 0;
  for (int t0 = 0; t0 < T; t0 += 32) {
    ((uint4*)(Ks[buf] + r * LSTR))[tid & 3] = rk;
    ((uint4*)(Vs[buf] + r * LSTR))[tid & 3] = rv;
    __syncthreads();
    if (t0 + 32 < T) {
      rk = *(const uint4*)(kT + rowbase + t0 + 32 + c);
      rv = *(const uint4*)(vT + rowbase + t0 + 32 + c);
    }
    F16x16 af, b0, b1;
    load_a_frag(af, Ks[buf] + (mt + (lane & 15)) * LSTR, kh);          // A = k^T : [d][t]
    load_b_frag(b0, Vs[buf] + (ntb + (lane & 15)) * LSTR, kh);         // B = v   : [t][l]
    load_b_frag(b1, Vs[buf] + (ntb + 16 + (lane & 15)) * LSTR, kh);
    acc[0] = __builtin_amdgcn_wmma_f32_16x16x32_bf16(false, af.v, false, b0.v, (short)0, acc[0], false, false);
    acc[1] = __builtin_amdgcn_wmma_f32_16x16x32_bf16(false, af.v, false, b1.v, (short)0, acc[1], false, false);
    buf ^= 1;
  }
#pragma unroll
  for (int j = 0; j < 2; ++j)
#pragma unroll
    for (int rr = 0; rr < 8; ++rr) {
      int dd = mt + rr + ((lane >= 16) ? 8 : 0);
      int ll = ntb + j * 16 + (lane & 15);
      AttT[((size_t)bh * 64 + ll) * 64 + dd] = (bf16)acc[j][rr];  // store transposed [l][d]
    }
}

// ---- y[b,:,h,:] = q_softmax @ att[b,h]; att tile fetched via TDM tensor_load_to_lds ----
__global__ __launch_bounds__(256) void ygemm_kernel(
    const bf16* __restrict__ Qsm, const bf16* __restrict__ AttT,
    float* __restrict__ Y, int T, int D, int H) {
  __shared__ __align__(16) bf16 Qs[128 * LSTR2];  // [t][dh], full K=64
  __shared__ __align__(16) bf16 Bs[64 * LSTR2];   // [l][d] tile, full K=64
  int bh = blockIdx.y;
  int b = bh / H, h = bh - b * H;
  int t0 = blockIdx.x * 128;
  int tid = threadIdx.x, lane = tid & 31, wave = tid >> 5;
  const int kh = (lane >= 16) ? 1 : 0;
  // stage Q tile: 128 rows x 8 uint4
#pragma unroll
  for (int it = 0; it < 4; ++it) {
    int u = tid + it * 256;
    int r = u >> 3, c = u & 7;
    ((uint4*)(Qs + r * LSTR2))[c] =
        *(const uint4*)(Qsm + ((size_t)(b * T + t0 + r)) * D + h * 64 + c * 8);
  }
#if USE_TDM
  // Tensor Data Mover: copy 64x64 bf16 att tile -> LDS with per-row padding
  // (pad_interval=32 DWORDs (code 4), pad_amount=4 DWORDs (code 3) => 72-elem rows).
  if (wave == 0) {
    unsigned lds_addr = (unsigned)(size_t)(&Bs[0]);
    unsigned long long ga = (unsigned long long)(size_t)(AttT + (size_t)bh * 4096);
    u32x4 g0;
    g0.x = 1u;                                    // count=1, user desc
    g0.y = lds_addr;                              // lds_addr
    g0.z = (unsigned)(ga & 0xffffffffu);          // global_addr[31:0]
    g0.w = (unsigned)((ga >> 32) & 0x01ffffffu) | 0x80000000u;  // addr[56:32] | type=2
    i32x8 g1;
    g1[0] = (int)((1u << 16) | (1u << 20) | (4u << 22) | (3u << 25)); // 2B elems, pad en
    g1[1] = (int)(64u << 16);   // tensor_dim0 = 64
    g1[2] = (int)(64u << 16);   // tensor_dim1 = 64
    g1[3] = (int)(64u << 16);   // tile_dim0 = 64
    g1[4] = 64;                 // tile_dim1 = 64
    g1[5] = 64;                 // tensor_dim0_stride = 64
    g1[6] = 0;
    g1[7] = 0;
    asm volatile("tensor_load_to_lds %0, %1" :: "s"(g0), "s"(g1) : "memory");
  }
  __builtin_amdgcn_s_wait_tensorcnt(0);
#else
#pragma unroll
  for (int it = 0; it < 2; ++it) {   // att^T tile: 64 rows x 8 uint4
    int u = tid + it * 256;
    int r = u >> 3, c = u & 7;
    ((uint4*)(Bs + r * LSTR2))[c] =
        *(const uint4*)(AttT + (size_t)bh * 4096 + r * 64 + c * 8);
  }
#endif
  __syncthreads();
  v8f acc[4] = {};
#pragma unroll
  for (int kk = 0; kk < 64; kk += 32) {
    F16x16 af, bfr[4];
    load_a_frag(af, Qs + (wave * 16 + (lane & 15)) * LSTR2 + kk, kh);
#pragma unroll
    for (int nt = 0; nt < 4; ++nt)
      load_b_frag(bfr[nt], Bs + (nt * 16 + (lane & 15)) * LSTR2 + kk, kh);
#pragma unroll
    for (int nt = 0; nt < 4; ++nt)
      acc[nt] = __builtin_amdgcn_wmma_f32_16x16x32_bf16(false, af.v, false, bfr[nt].v, (short)0, acc[nt], false, false);
  }
#pragma unroll
  for (int nt = 0; nt < 4; ++nt)
#pragma unroll
    for (int r = 0; r < 8; ++r) {
      int t = t0 + wave * 16 + r + ((lane >= 16) ? 8 : 0);
      int cl = h * 64 + nt * 16 + (lane & 15);
      Y[((size_t)(b * T + t)) * D + cl] = acc[nt][r];
    }
}

// ---- emb_out = silu(emb) @ emb_W + emb_b  (one block per output) ----
__global__ __launch_bounds__(256) void emb_gemm_kernel(
    const float* __restrict__ Emb, const float* __restrict__ W,
    const float* __restrict__ bias, float* __restrict__ SS, int TE, int N2) {
  int o = blockIdx.x;
  int b = o / N2, j = o - b * N2;
  float p = 0.f;
  for (int i = threadIdx.x; i < TE; i += 256) {
    float e = Emb[b * TE + i];
    float s = e / (1.0f + __expf(-e));
    p += s * W[(size_t)i * N2 + j];
  }
  p = blockReduceSum(p);
  if (threadIdx.x == 0) SS[o] = p + bias[j];
}

// ---- h = silu( ln(y)*(1+scale) + shift ) -> bf16 ----
__global__ __launch_bounds__(256) void ln2_mod_kernel(
    const float* __restrict__ Y, const float* __restrict__ g, const float* __restrict__ bt,
    const float* __restrict__ SS, bf16* __restrict__ Out, int T, int D) {
  size_t row = blockIdx.x;
  int b = blockIdx.x / T;
  const float* yr = Y + row * D;
  float vals[4]; float s = 0.f;
#pragma unroll
  for (int i = 0; i < 4; ++i) { vals[i] = yr[threadIdx.x + i * 256]; s += vals[i]; }
  float mean = blockReduceSum(s) * (1.0f / D);
  float sq = 0.f;
#pragma unroll
  for (int i = 0; i < 4; ++i) { float d = vals[i] - mean; sq += d * d; }
  float rstd = rsqrtf(blockReduceSum(sq) * (1.0f / D) + 1e-5f);
#pragma unroll
  for (int i = 0; i < 4; ++i) {
    int c = threadIdx.x + i * 256;
    float hn = (vals[i] - mean) * rstd * g[c] + bt[c];
    float h = hn * (1.0f + SS[b * 2 * D + c]) + SS[b * 2 * D + D + c];
    h = h / (1.0f + __expf(-h));
    Out[row * D + c] = (bf16)h;
  }
}

extern "C" void kernel_launch(void* const* d_in, const int* in_sizes, int n_in,
                              void* d_out, int out_size, void* d_ws, size_t ws_size,
                              hipStream_t stream) {
  constexpr int B = 4, T = 2048, D = 1024, H = 16, TE = 2048;
  constexpr int M = B * T;  // 8192
  const float* x        = (const float*)d_in[0];
  const float* emb      = (const float*)d_in[1];
  const float* src_mask = (const float*)d_in[2];
  const float* ln_w  = (const float*)d_in[3];
  const float* ln_b  = (const float*)d_in[4];
  const float* Wq    = (const float*)d_in[5];
  const float* bq    = (const float*)d_in[6];
  const float* Wk    = (const float*)d_in[7];
  const float* bk    = (const float*)d_in[8];
  const float* Wv    = (const float*)d_in[9];
  const float* bv    = (const float*)d_in[10];
  const float* emb_W = (const float*)d_in[11];
  const float* emb_b = (const float*)d_in[12];
  const float* ln2_w = (const float*)d_in[13];
  const float* ln2_b = (const float*)d_in[14];
  const float* out_W = (const float*)d_in[15];
  const float* out_b = (const float*)d_in[16];
  float* out = (float*)d_out;
  (void)in_sizes; (void)n_in; (void)out_size; (void)ws_size;

  char* ws = (char*)d_ws;
  size_t off = 0;
  auto alloc = [&](size_t bytes) { void* p = ws + off; off += (bytes + 255) & ~(size_t)255; return p; };

  bf16*  xn   = (bf16*)alloc((size_t)M * D * 2);
  bf16*  wq_t = (bf16*)alloc((size_t)D * D * 2);   // transposed bf16 weights (N x K)
  bf16*  wk_t = (bf16*)alloc((size_t)D * D * 2);
  bf16*  wv_t = (bf16*)alloc((size_t)D * D * 2);
  bf16*  wo_t = (bf16*)alloc((size_t)D * D * 2);
  float* bufQ = (float*)alloc((size_t)M * D * 4);  // q raw, reused for y
  float* bufK = (float*)alloc((size_t)M * D * 4);  // k raw, reused for hs (bf16)
  bf16*  q_b  = (bf16*)alloc((size_t)M * D * 2);   // q softmax (token-major)
  bf16*  kTm  = (bf16*)alloc((size_t)M * D * 2);   // k softmax, time-major [b,d,t]
  bf16*  vTm  = (bf16*)alloc((size_t)M * D * 2);   // v masked, time-major [b,d,t]
  bf16*  attT = (bf16*)alloc((size_t)B * H * 64 * 64 * 2);  // transposed [l][d]
  float* ss   = (float*)alloc((size_t)B * 2 * D * 4);

  // 1) convert + transpose weights to bf16 (N x K)
  f32_to_bf16_T_kernel<<<1024, 256, 0, stream>>>(Wq, wq_t, D, D);
  f32_to_bf16_T_kernel<<<1024, 256, 0, stream>>>(Wk, wk_t, D, D);
  f32_to_bf16_T_kernel<<<1024, 256, 0, stream>>>(Wv, wv_t, D, D);
  f32_to_bf16_T_kernel<<<1024, 256, 0, stream>>>(out_W, wo_t, D, D);
  // 2) LN1 -> bf16 activations
  layernorm_bf16_kernel<<<M, 256, 0, stream>>>(x, ln_w, ln_b, xn, D);
  // 3) Q/K/V GEMMs (WMMA, 128x256 block tile)
  dim3 gg(D / 256, M / 128);
  gemm_bf16_kernel<0><<<gg, 256, 0, stream>>>(xn, wq_t, bq, bufQ, nullptr, nullptr, M, D, D, T);
  gemm_bf16_kernel<0><<<gg, 256, 0, stream>>>(xn, wk_t, bk, bufK, nullptr, nullptr, M, D, D, T);
  gemm_bf16_kernel<1><<<gg, 256, 0, stream>>>(xn, wv_t, bv, vTm, src_mask, nullptr, M, D, D, T);
  // 4) softmaxes
  q_softmax_kernel<<<M * H / 8, 256, 0, stream>>>(bufQ, q_b);
  time_softmax_kernel<<<B * D, 256, 0, stream>>>(bufK, src_mask, kTm, T, D);
  // 5) linear attention: state then projection (WMMA)
  attn_state_kernel<<<B * H, 256, 0, stream>>>(kTm, vTm, attT, T, D, H);
  float* y = bufQ;
  ygemm_kernel<<<dim3(T / 128, B * H), 256, 0, stream>>>(q_b, attT, y, T, D, H);
  // 6) stylization block
  emb_gemm_kernel<<<B * 2 * D, 256, 0, stream>>>(emb, emb_W, emb_b, ss, TE, 2 * D);
  bf16* hs = (bf16*)bufK;
  ln2_mod_kernel<<<M, 256, 0, stream>>>(y, ln2_w, ln2_b, ss, hs, T, D);
  // 7) final GEMM with fused residual: out = x + hs @ out_W + out_b
  gemm_bf16_kernel<2><<<gg, 256, 0, stream>>>(hs, wo_t, out_b, out, nullptr, x, M, D, D, T);
}
